// FA3MLAAttention_45140106281224
// MI455X (gfx1250) — compile-verified, matrix operand order checked by reading, not run on hardware
//
#include <hip/hip_runtime.h>

// ---------------------------------------------------------------------------
// MLA attention for MI455X (gfx1250), wave32, bf16 WMMA (16x16x32), fp32 acc.
// High-density tiles: GEMM wave tile 64x64 (16 WMMA/step), attention wave
// tile 32 q-rows.  Double-buffered LDS, global_prefetch, fused flash attn.
// ---------------------------------------------------------------------------

typedef __attribute__((ext_vector_type(16))) __bf16 v16bf;
typedef __attribute__((ext_vector_type(8)))  __bf16 v8bf;
typedef __attribute__((ext_vector_type(4)))  __bf16 v4bf;
typedef __attribute__((ext_vector_type(8)))  float  v8f;
typedef __attribute__((ext_vector_type(4)))  float  v4f;

#define WMMA_BF16(a, b, c) \
  __builtin_amdgcn_wmma_f32_16x16x32_bf16(false, (a), false, (b), (short)0, (c), false, false)

static __device__ __forceinline__ __bf16 f2bf(float f) {
  union { float f; unsigned u; } in; in.f = f;
  unsigned r = in.u + 0x7fffu + ((in.u >> 16) & 1u);   // round-to-nearest-even
  union { unsigned short s; __bf16 b; } out; out.s = (unsigned short)(r >> 16);
  return out.b;
}
static __device__ __forceinline__ float bf2f(__bf16 b) {
  union { unsigned short s; __bf16 b; } in; in.b = b;
  union { unsigned u; float f; } out; out.u = ((unsigned)in.s) << 16;
  return out.f;
}

// 16x32 bf16 A-fragment (or N-major B^T fragment): lane L row = L&15,
// K = ko..ko+7 and ko+16..ko+23, ko = (L>=16)*8.  Two 16B LDS loads.
static __device__ __forceinline__ v16bf ldfrag(const __bf16* p) {
  v8bf lo = *(const v8bf*)(p);
  v8bf hi = *(const v8bf*)(p + 16);
  return __builtin_shufflevector(lo, hi, 0,1,2,3,4,5,6,7,8,9,10,11,12,13,14,15);
}

// ---------------------------------------------------------------------------
// fp32 -> bf16 cast, 4 elems/thread
// ---------------------------------------------------------------------------
__global__ __launch_bounds__(256) void cast_f32_bf16(const float* __restrict__ in,
                                                     __bf16* __restrict__ out, int n4) {
  int i = blockIdx.x * 256 + threadIdx.x;
  if (i >= n4) return;
  v4f x = *(const v4f*)(in + (size_t)i * 4);
  v4bf y;
  y[0] = f2bf(x[0]); y[1] = f2bf(x[1]); y[2] = f2bf(x[2]); y[3] = f2bf(x[3]);
  *(v4bf*)(out + (size_t)i * 4) = y;
}

// ---------------------------------------------------------------------------
// bf16 GEMM, compile-time N/K.  C[M,N] = A[M,K]*B[K,N].  Block 128x256, BK=32,
// 256 threads = 8 waves in 2(m) x 4(n); wave tile 64x64 = 4x4 WMMA frags
// (16 WMMA per K-step, 8 fragment loads).  Register-staged LDS double buffer:
// one barrier per K-step, next tile's global loads issued before compute,
// L2 prefetch (global_prefetch_b8) two tiles ahead.
// ---------------------------------------------------------------------------
#define BM 128
#define BN 256
#define BK 32

template <int N, int K, bool OUTF32>
__global__ __launch_bounds__(256) void gemm_bf16(const __bf16* __restrict__ A,
                                                 const __bf16* __restrict__ B,
                                                 void* __restrict__ Cp) {
  __shared__ __bf16 As[2][BM * BK];   // 2 x  8 KB (M x K, row-major)
  __shared__ __bf16 Bs[2][BN * BK];   // 2 x 16 KB (N x K, transposed)

  const int m0 = blockIdx.y * BM;
  const int n0 = blockIdx.x * BN;
  const int tid  = threadIdx.x;
  const int lane = tid & 31;
  const int wid  = tid >> 5;
  const int wm   = wid & 1;    // 64-row group
  const int wn   = wid >> 1;   // 64-col group
  const int hrow = lane & 15;
  const int hf   = lane >> 4;
  const int ko   = hf * 8;

  // staging assignments
  const int arow = tid >> 1;           // 0..127
  const int acol = (tid & 1) << 4;     // 0 / 16
  const int brow = tid & 31;           // k row
  const int bcol = (tid >> 5) << 5;    // n group of 32
  const __bf16* Ag = A + (size_t)(m0 + arow) * K + acol;
  const __bf16* Bg = B + (size_t)brow * N + n0 + bcol;

  v8f acc[4][4] = {};
  v8bf a0, a1, b0, b1, b2, b3;

  // preload tile 0 and stage into buffer 0
  a0 = *(const v8bf*)(Ag);      a1 = *(const v8bf*)(Ag + 8);
  b0 = *(const v8bf*)(Bg);      b1 = *(const v8bf*)(Bg + 8);
  b2 = *(const v8bf*)(Bg + 16); b3 = *(const v8bf*)(Bg + 24);
  *(v8bf*)(&As[0][arow * BK + acol])     = a0;
  *(v8bf*)(&As[0][arow * BK + acol + 8]) = a1;
#pragma unroll
  for (int j = 0; j < 8; ++j) {
    Bs[0][(bcol + j)      * BK + brow] = b0[j];
    Bs[0][(bcol + 8 + j)  * BK + brow] = b1[j];
    Bs[0][(bcol + 16 + j) * BK + brow] = b2[j];
    Bs[0][(bcol + 24 + j) * BK + brow] = b3[j];
  }
  __syncthreads();

  const int nk = K / BK;
  for (int kt = 0; kt < nk; ++kt) {
    const int cur = kt & 1;
    const int nxt = cur ^ 1;

    // issue global loads for tile kt+1 (overlapped with WMMA below)
    if (kt + 1 < nk) {
      const __bf16* ag = Ag + (kt + 1) * BK;
      const __bf16* bg = Bg + (size_t)(kt + 1) * BK * N;
      a0 = *(const v8bf*)(ag);      a1 = *(const v8bf*)(ag + 8);
      b0 = *(const v8bf*)(bg);      b1 = *(const v8bf*)(bg + 8);
      b2 = *(const v8bf*)(bg + 16); b3 = *(const v8bf*)(bg + 24);
    }
    if (kt + 2 < nk) {   // L2 prefetch of tile kt+2
      __builtin_prefetch(Ag + (kt + 2) * BK, 0, 1);
      __builtin_prefetch(Bg + (size_t)(kt + 2) * BK * N, 0, 1);
    }

    // compute tile kt from LDS buffer `cur`: 8 fragment loads, 16 WMMAs
    v16bf af[4], bfr[4];
#pragma unroll
    for (int i = 0; i < 4; ++i)
      af[i] = ldfrag(&As[cur][(wm * 64 + i * 16 + hrow) * BK + ko]);
#pragma unroll
    for (int j = 0; j < 4; ++j)
      bfr[j] = ldfrag(&Bs[cur][(wn * 64 + j * 16 + hrow) * BK + ko]);
#pragma unroll
    for (int i = 0; i < 4; ++i)
#pragma unroll
      for (int j = 0; j < 4; ++j)
        acc[i][j] = WMMA_BF16(af[i], bfr[j], acc[i][j]);

    // stage registers into buffer `nxt` (not read this iteration)
    if (kt + 1 < nk) {
      *(v8bf*)(&As[nxt][arow * BK + acol])     = a0;
      *(v8bf*)(&As[nxt][arow * BK + acol + 8]) = a1;
#pragma unroll
      for (int j = 0; j < 8; ++j) {
        Bs[nxt][(bcol + j)      * BK + brow] = b0[j];
        Bs[nxt][(bcol + 8 + j)  * BK + brow] = b1[j];
        Bs[nxt][(bcol + 16 + j) * BK + brow] = b2[j];
        Bs[nxt][(bcol + 24 + j) * BK + brow] = b3[j];
      }
    }
    __syncthreads();
  }

  // epilogue: base pointer per fragment, 8 stores at constant offsets e*N
#pragma unroll
  for (int i = 0; i < 4; ++i)
#pragma unroll
    for (int j = 0; j < 4; ++j) {
      const int row0 = m0 + wm * 64 + i * 16 + hf * 8;
      const int col  = n0 + wn * 64 + j * 16 + hrow;
      if (OUTF32) {
        float* cp = (float*)Cp + (size_t)row0 * N + col;
#pragma unroll
        for (int e = 0; e < 8; ++e) cp[e * N] = acc[i][j][e];
      } else {
        __bf16* cp = (__bf16*)Cp + (size_t)row0 * N + col;
#pragma unroll
        for (int e = 0; e < 8; ++e) cp[e * N] = f2bf(acc[i][j][e]);
      }
    }
}

// ---------------------------------------------------------------------------
// RoPE in-place on the rope slice [128..192) of each head.  X: (B*T, 3072).
// ---------------------------------------------------------------------------
__global__ __launch_bounds__(256) void rope_kernel(__bf16* __restrict__ X,
                                                   const float* __restrict__ freqs,
                                                   const int* __restrict__ start_pos,
                                                   int rows, int T) {
  int gid = blockIdx.x * 256 + threadIdx.x;
  int total = rows * 16 * 32;
  if (gid >= total) return;
  int j   = gid & 31;
  int h   = (gid >> 5) & 15;
  int row = gid >> 9;
  int t   = (row % T) + start_pos[0];
  float c = freqs[t * 64 + j * 2 + 0];
  float s = freqs[t * 64 + j * 2 + 1];
  size_t base = (size_t)row * 3072 + h * 192 + 128;
  float x1 = bf2f(X[base + j]);
  float x2 = bf2f(X[base + j + 32]);
  X[base + j]      = f2bf(x1 * c - x2 * s);
  X[base + j + 32] = f2bf(x2 * c + x1 * s);
}

// ---------------------------------------------------------------------------
// Fused flash attention.  Block = (b, h, 128-row q tile), 4 waves, each wave
// owns 32 q rows (2 row-frags).  Key tile 32.  S: 24 WMMA/tile at 1:1
// load:wmma ratio, online softmax via shfl_xor butterflies over 16-lane
// halves, P restaged through LDS to A-layout, PV: 16 WMMA/tile.
// ---------------------------------------------------------------------------
#define HD 192
#define VD 128
#define TT 2048
__global__ __launch_bounds__(128) void mla_attn(const __bf16* __restrict__ Q,
                                                const __bf16* __restrict__ Kb,
                                                const __bf16* __restrict__ Vb,
                                                __bf16* __restrict__ O) {
  __shared__ __bf16 Qs[128 * HD];      // 48 KB
  __shared__ __bf16 Ks[32 * HD];       // 12 KB   (key, d)
  __shared__ __bf16 Vts[VD * 32];      //  8 KB   (d, key) = V^T tile
  __shared__ __bf16 Ps[4][32 * 32];    //  8 KB   per-wave P tile (32 rows)

  const float SCALE = 0.07216878364870322f;   // 1/sqrt(192)
  const int qt = blockIdx.x;
  const int b  = blockIdx.y >> 4;
  const int h  = blockIdx.y & 15;
  const int tid  = threadIdx.x;
  const int lane = tid & 31;
  const int wv   = tid >> 5;
  const int hrow = lane & 15;
  const int hf   = lane >> 4;
  const int ko   = hf * 8;

  // ---- stage Q tile (128 x 192): one row per thread ----
  {
    const __bf16* g = Q + ((size_t)(b * TT + qt * 128 + tid) * 3072 + h * HD);
    __bf16* d = Qs + tid * HD;
#pragma unroll
    for (int u = 0; u < 24; ++u)
      *(v8bf*)(d + u * 8) = *(const v8bf*)(g + u * 8);
  }
  __syncthreads();

  float m_r[2][8], l_r[2][8];
#pragma unroll
  for (int r = 0; r < 2; ++r)
#pragma unroll
    for (int e = 0; e < 8; ++e) { m_r[r][e] = -1e30f; l_r[r][e] = 0.0f; }
  v8f Oacc[2][8] = {};

  const int nkt = qt * 4 + 4;
  for (int kt = 0; kt < nkt; ++kt) {
    // ---- stage K tile (32 x 192) ----
    {
      int r     = tid & 31;
      int cbase = (tid >> 5) * 48;
      const __bf16* g = Kb + ((size_t)(b * TT + kt * 32 + r) * 3072 + h * HD + cbase);
      if (kt + 1 < nkt)
        __builtin_prefetch(g + (size_t)32 * 3072, 0, 1);
      __bf16* d = Ks + r * HD + cbase;
#pragma unroll
      for (int u = 0; u < 6; ++u)
        *(v8bf*)(d + u * 8) = *(const v8bf*)(g + u * 8);
    }
    // ---- stage V tile transposed (d, key) ----
    {
      int r     = tid & 31;
      int dbase = (tid >> 5) * 32;
      const __bf16* g = Vb + ((size_t)(b * TT + kt * 32 + r) * 2048 + h * VD + dbase);
      if (kt + 1 < nkt)
        __builtin_prefetch(g + (size_t)32 * 2048, 0, 1);
      v8bf v0 = *(const v8bf*)(g);
      v8bf v1 = *(const v8bf*)(g + 8);
      v8bf v2 = *(const v8bf*)(g + 16);
      v8bf v3 = *(const v8bf*)(g + 24);
#pragma unroll
      for (int j = 0; j < 8; ++j) {
        Vts[(dbase + j)      * 32 + r] = v0[j];
        Vts[(dbase + 8 + j)  * 32 + r] = v1[j];
        Vts[(dbase + 16 + j) * 32 + r] = v2[j];
        Vts[(dbase + 24 + j) * 32 + r] = v3[j];
      }
    }
    __syncthreads();

    // ---- S = Q (32x192) * K^T (192x32): 4 frag loads per 4 WMMAs ----
    v8f S[2][2] = {};
#pragma unroll
    for (int dc = 0; dc < 6; ++dc) {
      v16bf aq0 = ldfrag(Qs + (wv * 32 + hrow)      * HD + dc * 32 + ko);
      v16bf aq1 = ldfrag(Qs + (wv * 32 + 16 + hrow) * HD + dc * 32 + ko);
      v16bf bk0 = ldfrag(Ks + (hrow)      * HD + dc * 32 + ko);
      v16bf bk1 = ldfrag(Ks + (16 + hrow) * HD + dc * 32 + ko);
      S[0][0] = WMMA_BF16(aq0, bk0, S[0][0]);
      S[0][1] = WMMA_BF16(aq0, bk1, S[0][1]);
      S[1][0] = WMMA_BF16(aq1, bk0, S[1][0]);
      S[1][1] = WMMA_BF16(aq1, bk1, S[1][1]);
    }

    // ---- scale, causal mask, online softmax ----
    const int key0 = kt * 32 + hrow;
    const int key1 = key0 + 16;
#pragma unroll
    for (int r = 0; r < 2; ++r) {
      const int qrow0 = qt * 128 + wv * 32 + r * 16 + hf * 8;
#pragma unroll
      for (int e = 0; e < 8; ++e) {
        int qr = qrow0 + e;
        float s0 = S[r][0][e] * SCALE; if (key0 > qr) s0 = -1e30f;
        float s1 = S[r][1][e] * SCALE; if (key1 > qr) s1 = -1e30f;
        float mx = fmaxf(s0, s1);
        mx = fmaxf(mx, __shfl_xor(mx, 1, 32));
        mx = fmaxf(mx, __shfl_xor(mx, 2, 32));
        mx = fmaxf(mx, __shfl_xor(mx, 4, 32));
        mx = fmaxf(mx, __shfl_xor(mx, 8, 32));
        float mnew = fmaxf(m_r[r][e], mx);
        float p0 = __expf(s0 - mnew);
        float p1 = __expf(s1 - mnew);
        float rs = p0 + p1;
        rs += __shfl_xor(rs, 1, 32);
        rs += __shfl_xor(rs, 2, 32);
        rs += __shfl_xor(rs, 4, 32);
        rs += __shfl_xor(rs, 8, 32);
        float fac = __expf(m_r[r][e] - mnew);
        l_r[r][e] = l_r[r][e] * fac + rs;
        m_r[r][e] = mnew;
#pragma unroll
        for (int nd = 0; nd < 8; ++nd) Oacc[r][nd][e] *= fac;
        Ps[wv][(r * 16 + e + hf * 8) * 32 + hrow]      = f2bf(p0);
        Ps[wv][(r * 16 + e + hf * 8) * 32 + 16 + hrow] = f2bf(p1);
      }
    }
#if __has_builtin(__builtin_amdgcn_wave_barrier)
    __builtin_amdgcn_wave_barrier();
#endif

    // ---- O += P (32x32) * V (32x128): 10 frag loads per 16 WMMAs ----
    v16bf ap0 = ldfrag(Ps[wv] + (hrow)      * 32 + ko);
    v16bf ap1 = ldfrag(Ps[wv] + (16 + hrow) * 32 + ko);
#pragma unroll
    for (int nd = 0; nd < 8; ++nd) {
      v16bf bv = ldfrag(Vts + (nd * 16 + hrow) * 32 + ko);
      Oacc[0][nd] = WMMA_BF16(ap0, bv, Oacc[0][nd]);
      Oacc[1][nd] = WMMA_BF16(ap1, bv, Oacc[1][nd]);
    }
    __syncthreads();
  }

  // ---- epilogue ----
#pragma unroll
  for (int r = 0; r < 2; ++r)
#pragma unroll
    for (int e = 0; e < 8; ++e) {
      float inv = 1.0f / l_r[r][e];
      int grow = b * TT + qt * 128 + wv * 32 + r * 16 + hf * 8 + e;
      __bf16* op = O + (size_t)grow * 2048 + h * VD + hrow;
#pragma unroll
      for (int nd = 0; nd < 8; ++nd)
        op[nd * 16] = f2bf(Oacc[r][nd][e] * inv);
    }
}

// ---------------------------------------------------------------------------
// Host-side orchestration
// ---------------------------------------------------------------------------
extern "C" void kernel_launch(void* const* d_in, const int* in_sizes, int n_in,
                              void* d_out, int out_size, void* d_ws, size_t ws_size,
                              hipStream_t stream) {
  (void)in_sizes; (void)n_in; (void)out_size; (void)ws_size;
  const int Bb = 2, T = 2048, C = 2048, H = 16;
  const int HDh = 192, VDh = 128, RK = 256;
  const int M  = Bb * T;          // 4096
  const int NQ = H * HDh;         // 3072
  const int NV = H * VDh;         // 2048

  const float* x     = (const float*)d_in[0];
  const float* freqs = (const float*)d_in[1];
  const float* w_q   = (const float*)d_in[2];
  const float* w_kv  = (const float*)d_in[3];
  const float* w_ku  = (const float*)d_in[4];
  const float* w_vu  = (const float*)d_in[5];
  const float* w_o   = (const float*)d_in[6];
  const int*   spos  = (const int*)d_in[7];

  char* ws = (char*)d_ws;
  size_t off = 0;
  auto alloc = [&](size_t elems) {
    size_t o = off;
    off += (elems * sizeof(__bf16) + 255) & ~(size_t)255;
    return (__bf16*)(ws + o);
  };
  __bf16* xb   = alloc((size_t)M * C);
  __bf16* wqb  = alloc((size_t)C * NQ);
  __bf16* wkvb = alloc((size_t)C * RK);
  __bf16* wkub = alloc((size_t)RK * NQ);
  __bf16* wvub = alloc((size_t)RK * NV);
  __bf16* wob  = alloc((size_t)NV * C);
  __bf16* qb   = alloc((size_t)M * NQ);
  __bf16* kvb  = alloc((size_t)M * RK);
  __bf16* kb   = alloc((size_t)M * NQ);
  __bf16* vb   = alloc((size_t)M * NV);
  __bf16* aob  = alloc((size_t)M * NV);

  auto cast = [&](const float* src, __bf16* dst, int n) {
    int n4 = n / 4;
    cast_f32_bf16<<<(n4 + 255) / 256, 256, 0, stream>>>(src, dst, n4);
  };
  cast(x,    xb,   M * C);
  cast(w_q,  wqb,  C * NQ);
  cast(w_kv, wkvb, C * RK);
  cast(w_ku, wkub, RK * NQ);
  cast(w_vu, wvub, RK * NV);
  cast(w_o,  wob,  NV * C);

  // projections (N, K compile-time); block tile 128x256
  gemm_bf16<3072, 2048, false><<<dim3(NQ / BN, M / BM), 256, 0, stream>>>(xb,  wqb,  (void*)qb);
  gemm_bf16< 256, 2048, false><<<dim3(RK / BN, M / BM), 256, 0, stream>>>(xb,  wkvb, (void*)kvb);
  gemm_bf16<3072,  256, false><<<dim3(NQ / BN, M / BM), 256, 0, stream>>>(kvb, wkub, (void*)kb);
  gemm_bf16<2048,  256, false><<<dim3(NV / BN, M / BM), 256, 0, stream>>>(kvb, wvub, (void*)vb);

  // rotary embedding on q and k
  int rope_total = M * H * 32;
  rope_kernel<<<(rope_total + 255) / 256, 256, 0, stream>>>(qb, freqs, spos, M, T);
  rope_kernel<<<(rope_total + 255) / 256, 256, 0, stream>>>(kb, freqs, spos, M, T);

  // fused flash attention (128-row q tiles)
  mla_attn<<<dim3(T / 128, Bb * H), 128, 0, stream>>>(qb, kb, vb, aob);

  // output projection -> fp32
  gemm_bf16<2048, 2048, true><<<dim3(C / BN, M / BM), 256, 0, stream>>>(aob, wob, d_out);
}
